// Model0_6305011990976
// MI455X (gfx1250) — matrix operand order repaired and checked by reading, not compiled, not run on hardware
//
#include <hip/hip_runtime.h>

// ---------------------------------------------------------------------------
// 3-layer decayed GRU on MI455X (gfx1250, wave32, WMMA bf16 16x16x32).
// B=128, T=256, IN=128, H=512, C=16.
//   * pack_w: fp32 weights -> bf16 WMMA B-fragment layout (per call).
//   * gru_layer: gi/gh GEMMs via v_wmma_f32_16x16x32_bf16 with named register
//     accumulators (no dynamic acc indexing -> no scratch spill), fused GRU
//     gate epilogue.
//   * out_proj: softmax head (tiny, fp32 VALU).
// Sequential recurrence = 4 launches/step, ping-pong fp32 h buffers in d_ws.
// ---------------------------------------------------------------------------

#define Bc   128
#define Tc   256
#define INc  128
#define Hc   512
#define Cc   16
#define PK   128   // K panel staged in LDS
#define PST  136   // LDS row stride (bf16 elems), 272B (16B aligned, even)

typedef __attribute__((ext_vector_type(16))) __bf16 v16bf;
typedef __attribute__((ext_vector_type(8)))  float  v8f;

union Frag32 {
    v16bf v;
    uint4 u[2];
};

static __device__ __forceinline__ unsigned short f2bf(float f) {
    unsigned int u = __float_as_uint(f);
    u += 0x7fffu + ((u >> 16) & 1u);           // round-to-nearest-even
    return (unsigned short)(u >> 16);
}

static __device__ __forceinline__ float sigmoidf_(float x) {
    return 1.0f / (1.0f + __expf(-x));
}

static __device__ __forceinline__ v8f wmma_bf16(const Frag32& a, const Frag32& b,
                                                const v8f& c) {
    return __builtin_amdgcn_wmma_f32_16x16x32_bf16(
        false, a.v, false, b.v, (short)0, c, false, false);
}

// ---------------------------------------------------------------------------
// Pack W [1536, K] row-major fp32 into bf16 B-fragments for 16x16x32 WMMA.
// B tile = 32x16 (KxN); fragment elem (lane L, i):
//   L 0..15 : N = L,      K = k0 + i
//   L 16..31: N = L - 16, K = k0 + 16 + i
// Fragment f = NT*(K/32) + kt ; element e = f*512 + L*16 + i.
// ---------------------------------------------------------------------------
__global__ void pack_w(const float* __restrict__ W,
                       unsigned short* __restrict__ dst, int K, int total) {
    for (int e = blockIdx.x * blockDim.x + threadIdx.x; e < total;
         e += gridDim.x * blockDim.x) {
        int frag = e >> 9;
        int r    = e & 511;
        int lane = r >> 4;
        int i    = r & 15;
        int fragsPerN = K >> 5;
        int NT = frag / fragsPerN;
        int kt = frag - NT * fragsPerN;
        int n  = (NT << 4) + (lane & 15);
        int k  = (kt << 5) + ((lane & 16) ? 16 : 0) + i;
        dst[e] = f2bf(W[n * K + k]);
    }
}

// ---------------------------------------------------------------------------
// One staged K panel (128 wide) of the three gate-chunk GEMMs.
// Accumulators are caller registers, bound by reference at compile time.
// ---------------------------------------------------------------------------
static __device__ __forceinline__ void gemm_panel(
    const unsigned short* __restrict__ Wp, const unsigned short* As,
    int fragsPerN, int p, int bx, int lane, int wid,
    v8f& aR0, v8f& aR1, v8f& aZ0, v8f& aZ1, v8f& aN0, v8f& aN1) {
    const int row = (wid << 4) + (lane & 15);
    const unsigned short* rp = &As[row * PST];
    const int la = (lane < 16) ? 0 : 16;
    const int lb = (lane < 16) ? 8 : 24;
    const int base = bx << 1;
#pragma unroll
    for (int ks = 0; ks < 4; ++ks) {
        const int kb = ks << 5;
        Frag32 fa;
        fa.u[0] = *(const uint4*)(rp + kb + la);
        fa.u[1] = *(const uint4*)(rp + kb + lb);
        const int kt = (p << 2) + ks;

        auto ldb = [&](int NT) {
            const unsigned short* fp =
                Wp + ((size_t)(NT * fragsPerN + kt) << 9) + (lane << 4);
            Frag32 fb;
            fb.u[0] = *(const uint4*)(fp);
            fb.u[1] = *(const uint4*)(fp + 8);
            return fb;
        };
        Frag32 b0 = ldb(base + 0);       aR0 = wmma_bf16(fa, b0, aR0);
        Frag32 b1 = ldb(base + 1);       aR1 = wmma_bf16(fa, b1, aR1);
        Frag32 b2 = ldb(32 + base + 0);  aZ0 = wmma_bf16(fa, b2, aZ0);
        Frag32 b3 = ldb(32 + base + 1);  aZ1 = wmma_bf16(fa, b3, aZ1);
        Frag32 b4 = ldb(64 + base + 0);  aN0 = wmma_bf16(fa, b4, aN0);
        Frag32 b5 = ldb(64 + base + 1);  aN1 = wmma_bf16(fa, b5, aN1);
    }
}

// Per-(ntile) GRU gate epilogue over the 8 rows this lane owns.
static __device__ __forceinline__ void epilogue_tile(
    int n, int wid, int lane, int t,
    const v8f& aR, const v8f& aZ, const v8f& aGiN, const v8f& aGhN,
    const float* __restrict__ b_ih, const float* __restrict__ b_hh,
    const float* __restrict__ dvec, const float* __restrict__ W_dec,
    const float* __restrict__ b_dec, const float* __restrict__ h_in,
    float* __restrict__ h_out) {
    const float bir  = b_ih[n] + b_hh[n];
    const float biz  = b_ih[Hc + n] + b_hh[Hc + n];
    const float bin_ = b_ih[2 * Hc + n];
    const float bhn  = b_hh[2 * Hc + n];
    const float wd = W_dec[n], bd = b_dec[n];
#pragma unroll
    for (int v = 0; v < 8; ++v) {
        const int m = (wid << 4) + ((lane < 16) ? v : v + 8);
        const float r  = sigmoidf_(aR[v] + bir);
        const float z  = sigmoidf_(aZ[v] + biz);
        const float ng = tanhf(aGiN[v] + bin_ + r * (aGhN[v] + bhn));
        float g = 1.0f;
        if (t != 0) g = __expf(-fmaxf(0.0f, dvec[m * Tc + t] * wd + bd));
        const float hd = h_in[m * Hc + n] * g;
        h_out[m * Hc + n] = (1.0f - z) * ng + z * hd;
    }
}

// ---------------------------------------------------------------------------
// One GRU layer, one timestep. Grid: 16 blocks x 256 threads (8 waves).
// Block owns 32 H-columns; wave w owns M-tile w (16 batch rows).
// ---------------------------------------------------------------------------
__global__ void __launch_bounds__(256)
gru_layer(const float* __restrict__ a_in, int in_is_x, int Kin, int t,
          const unsigned short* __restrict__ Wih_p,
          const unsigned short* __restrict__ Whh_p,
          const float* __restrict__ b_ih, const float* __restrict__ b_hh,
          const float* __restrict__ dvec, const float* __restrict__ W_dec,
          const float* __restrict__ b_dec,
          const float* __restrict__ h_in, float* __restrict__ h_out) {
    __shared__ __align__(16) unsigned short As[Bc * PST];

    const int tid  = threadIdx.x;
    const int lane = tid & 31;
    const int wid  = tid >> 5;                 // M-tile (0..7)
    const int bx   = blockIdx.x;
    const int c0   = bx * 32;                  // first H-column of this block

    const v8f zv = {0.f, 0.f, 0.f, 0.f, 0.f, 0.f, 0.f, 0.f};
    v8f aR0 = zv, aR1 = zv, aZ0 = zv, aZ1 = zv;
    v8f aGiN0 = zv, aGiN1 = zv, aGhN0 = zv, aGhN1 = zv;

    // -------- phase 0: gi = A_in @ W_ih^T --------
    {
        const int fragsPerN = Kin >> 5;
        const int npan      = Kin >> 7;
        for (int p = 0; p < npan; ++p) {
            __syncthreads();
            // stage 128x128 bf16 panel, packed-pair (b32) stores
            for (int i = tid; i < (Bc * PK) / 2; i += 256) {
                const int m  = i >> 6;
                const int k  = (i & 63) << 1;
                const int kg = (p << 7) + k;
                float v0, v1;
                if (in_is_x) {
                    const float* src = &a_in[(m * Tc + t) * INc + kg];
                    v0 = src[0]; v1 = src[1];
                } else {
                    const float* src = &a_in[m * Hc + kg];
                    v0 = src[0]; v1 = src[1];
                }
                const unsigned int pk =
                    (unsigned int)f2bf(v0) | ((unsigned int)f2bf(v1) << 16);
                *(unsigned int*)&As[m * PST + k] = pk;
            }
            __syncthreads();
            gemm_panel(Wih_p, As, fragsPerN, p, bx, lane, wid,
                       aR0, aR1, aZ0, aZ1, aGiN0, aGiN1);
        }
    }

    // -------- phase 1: gh = (h * gamma) @ W_hh^T --------
    {
        const int fragsPerN = Hc >> 5;         // 16
        for (int p = 0; p < 4; ++p) {
            __syncthreads();
            for (int i = tid; i < (Bc * PK) / 2; i += 256) {
                const int m  = i >> 6;
                const int k  = (i & 63) << 1;
                const int kg = (p << 7) + k;
                float g0 = 1.0f, g1 = 1.0f;
                if (t != 0) {
                    const float dd = dvec[m * Tc + t];
                    g0 = __expf(-fmaxf(0.0f, dd * W_dec[kg] + b_dec[kg]));
                    g1 = __expf(-fmaxf(0.0f, dd * W_dec[kg + 1] + b_dec[kg + 1]));
                }
                const float* src = &h_in[m * Hc + kg];
                const unsigned int pk =
                    (unsigned int)f2bf(src[0] * g0) |
                    ((unsigned int)f2bf(src[1] * g1) << 16);
                *(unsigned int*)&As[m * PST + k] = pk;
            }
            __syncthreads();
            gemm_panel(Whh_p, As, fragsPerN, p, bx, lane, wid,
                       aR0, aR1, aZ0, aZ1, aGhN0, aGhN1);
        }
    }

    // -------- fused GRU gate epilogue --------
    // C/D layout: elem (v, lane): M = (lane<16 ? v : v+8), N = lane&15.
    const int nl = lane & 15;
    epilogue_tile(c0 + nl,      wid, lane, t, aR0, aZ0, aGiN0, aGhN0,
                  b_ih, b_hh, dvec, W_dec, b_dec, h_in, h_out);
    epilogue_tile(c0 + 16 + nl, wid, lane, t, aR1, aZ1, aGiN1, aGhN1,
                  b_ih, b_hh, dvec, W_dec, b_dec, h_in, h_out);
}

// ---------------------------------------------------------------------------
// y_t = softmax(h3 @ W_out^T + b_out). Grid: 8 blocks x 256 thr; thread=(b,c).
// ---------------------------------------------------------------------------
__global__ void __launch_bounds__(256)
out_proj(const float* __restrict__ h3, const float* __restrict__ W_out,
         const float* __restrict__ b_out, float* __restrict__ y, int t) {
    __shared__ float lg[16][Cc + 1];
    const int tid = threadIdx.x;
    const int br  = blockIdx.x * 16 + (tid >> 4);
    const int c   = tid & 15;
    float acc = b_out[c];
    const float* hp = h3 + br * Hc;
    const float* wp = W_out + c * Hc;
#pragma unroll 4
    for (int k = 0; k < Hc; k += 4) {
        acc += hp[k] * wp[k] + hp[k + 1] * wp[k + 1] +
               hp[k + 2] * wp[k + 2] + hp[k + 3] * wp[k + 3];
    }
    lg[tid >> 4][c] = acc;
    __syncthreads();
    float mx = -INFINITY;
#pragma unroll
    for (int j = 0; j < Cc; ++j) mx = fmaxf(mx, lg[tid >> 4][j]);
    float s = 0.0f;
#pragma unroll
    for (int j = 0; j < Cc; ++j) s += __expf(lg[tid >> 4][j] - mx);
    y[(br * Tc + t) * Cc + c] = __expf(acc - mx) / s;
}

// ---------------------------------------------------------------------------
extern "C" void kernel_launch(void* const* d_in, const int* in_sizes, int n_in,
                              void* d_out, int out_size, void* d_ws,
                              size_t ws_size, hipStream_t stream) {
    const float* x     = (const float*)d_in[0];
    const float* dvec  = (const float*)d_in[1];
    const float* h10   = (const float*)d_in[2];
    const float* h20   = (const float*)d_in[3];
    const float* h30   = (const float*)d_in[4];
    const float* W_ih1 = (const float*)d_in[5];
    const float* W_hh1 = (const float*)d_in[6];
    const float* b_ih1 = (const float*)d_in[7];
    const float* b_hh1 = (const float*)d_in[8];
    const float* W_ih2 = (const float*)d_in[9];
    const float* W_hh2 = (const float*)d_in[10];
    const float* b_ih2 = (const float*)d_in[11];
    const float* b_hh2 = (const float*)d_in[12];
    const float* W_ih3 = (const float*)d_in[13];
    const float* W_hh3 = (const float*)d_in[14];
    const float* b_ih3 = (const float*)d_in[15];
    const float* b_hh3 = (const float*)d_in[16];
    const float* W_dec = (const float*)d_in[17];
    const float* b_dec = (const float*)d_in[18];
    const float* W_out = (const float*)d_in[19];
    const float* b_out = (const float*)d_in[20];
    float* y = (float*)d_out;

    // ---- carve workspace ----
    size_t off = 0;
    auto carve = [&](size_t bytes) -> char* {
        char* p = (char*)d_ws + off;
        off += (bytes + 255) & ~(size_t)255;
        return p;
    };
    unsigned short* wih1p = (unsigned short*)carve((size_t)1536 * INc * 2);
    unsigned short* whh1p = (unsigned short*)carve((size_t)1536 * Hc * 2);
    unsigned short* wih2p = (unsigned short*)carve((size_t)1536 * Hc * 2);
    unsigned short* whh2p = (unsigned short*)carve((size_t)1536 * Hc * 2);
    unsigned short* wih3p = (unsigned short*)carve((size_t)1536 * Hc * 2);
    unsigned short* whh3p = (unsigned short*)carve((size_t)1536 * Hc * 2);
    const size_t HB = (size_t)Bc * Hc * sizeof(float);
    float* h1b[2] = {(float*)carve(HB), (float*)carve(HB)};
    float* h2b[2] = {(float*)carve(HB), (float*)carve(HB)};
    float* h3b[2] = {(float*)carve(HB), (float*)carve(HB)};

    // ---- pack weights to bf16 WMMA fragments (deterministic, per call) ----
    {
        const float* Ws[6]    = {W_ih1, W_hh1, W_ih2, W_hh2, W_ih3, W_hh3};
        unsigned short* Ds[6] = {wih1p, whh1p, wih2p, whh2p, wih3p, whh3p};
        const int Ks[6]       = {INc, Hc, Hc, Hc, Hc, Hc};
        for (int i = 0; i < 6; ++i) {
            const int total = 1536 * Ks[i];
            pack_w<<<dim3((total + 255) / 256), dim3(256), 0, stream>>>(
                Ws[i], Ds[i], Ks[i], total);
        }
    }

    // ---- initial hidden states into ping-pong buffer 0 ----
    hipMemcpyAsync(h1b[0], h10, HB, hipMemcpyDeviceToDevice, stream);
    hipMemcpyAsync(h2b[0], h20, HB, hipMemcpyDeviceToDevice, stream);
    hipMemcpyAsync(h3b[0], h30, HB, hipMemcpyDeviceToDevice, stream);

    // ---- sequential recurrence: 4 launches per step ----
    for (int t = 0; t < Tc; ++t) {
        const int p = t & 1, q = p ^ 1;
        gru_layer<<<dim3(16), dim3(256), 0, stream>>>(
            x, 1, INc, t, wih1p, whh1p, b_ih1, b_hh1,
            dvec, W_dec, b_dec, h1b[p], h1b[q]);
        gru_layer<<<dim3(16), dim3(256), 0, stream>>>(
            h1b[q], 0, Hc, t, wih2p, whh2p, b_ih2, b_hh2,
            dvec, W_dec, b_dec, h2b[p], h2b[q]);
        gru_layer<<<dim3(16), dim3(256), 0, stream>>>(
            h2b[q], 0, Hc, t, wih3p, whh3p, b_ih3, b_hh3,
            dvec, W_dec, b_dec, h3b[p], h3b[q]);
        out_proj<<<dim3(8), dim3(256), 0, stream>>>(h3b[q], W_out, b_out, y, t);
    }
}
